// GraphAttentionLayer_60017872994887
// MI455X (gfx1250) — matrix operand order repaired and checked by reading, not compiled, not run on hardware
//
#include <hip/hip_runtime.h>

// ---------------------------------------------------------------------------
// GAT layer, MI455X (gfx1250, wave32).
// Roofline: adj (8192^2 int32 = 256 MB, read exactly once) dominates:
// ~11 us @ 23.3 TB/s. Never materialize the 8192x8192 attention matrix.
// Fused flash-softmax + v_wmma_f32_16x16x32_f16 for P@Wh; Wh (f16, 2 MB)
// stays L2-resident (192 MB L2) across all 512 row-blocks.
// ---------------------------------------------------------------------------

#define NN     8192
#define FIN    256
#define FOUT   128
#define ALPHA  0.2f
#define NEGINF (-9.0e15f)

typedef __attribute__((ext_vector_type(16))) _Float16 v16h;
typedef __attribute__((ext_vector_type(8)))  float    v8f;
typedef __attribute__((ext_vector_type(8)))  int      v8i;
typedef __attribute__((ext_vector_type(8)))  float    f32x8;

static __device__ __forceinline__ _Float16 f2h(float f) { return (_Float16)f; }
static __device__ __forceinline__ float    h2f(_Float16 x) { return (float)x; }

// ---------------------------------------------------------------------------
// Kernel 0: one-time W -> f16 hi/lo split, pre-swizzled into the WMMA
// B-matrix lane layout so kernel 1 does a single contiguous 32B load per
// (kc,nc) tile. Layout: [kc][nc][lane][t], t=0..15, K = kc*32 + (t<8 ? 8*hi+t : 16+8*hi+t-8),
// N = nc*16 + (lane&15).
// ---------------------------------------------------------------------------
__global__ __launch_bounds__(256) void gat_wprep_kernel(
    const float* __restrict__ W, _Float16* __restrict__ Whi,
    _Float16* __restrict__ Wlo) {
  const int idx = blockIdx.x * 256 + threadIdx.x;   // 0 .. 32767
  if (idx >= 8 * 8 * 32 * 16) return;
  const int t    = idx & 15;
  const int lane = (idx >> 4) & 31;
  const int nc   = (idx >> 9) & 7;
  const int kc   = (idx >> 12) & 7;
  const int hi = lane >> 4, ml = lane & 15;
  const float wv = W[(size_t)(kc * 32 + (t < 8 ? 8 * hi + t : 16 + 8 * hi + (t - 8))) * FOUT + nc * 16 + ml];
  const _Float16 bh = f2h(wv);
  Whi[idx] = bh;
  Wlo[idx] = f2h(wv - h2f(bh));
}

// ---------------------------------------------------------------------------
// Kernel 1: Wh = h @ W (16-row strip per wave), f16 hi/lo split (11+11
// mantissa bits ~ fp32) with 3 WMMAs per 32-wide K chunk. Emits Wh (f32)
// and WhT16 (f16, transposed) so kernel 3's B tiles are two contiguous 16B loads per lane.
// A layout (16-bit 16x32): lane<16 holds row=lane, K={0..7,16..23};
// lane>=16 holds K={8..15,24..31}. C: row = v + 8*hi, col = lane&15.
// ---------------------------------------------------------------------------
__global__ __launch_bounds__(32) void gat_wh_kernel(
    const float* __restrict__ h, const _Float16* __restrict__ Whi,
    const _Float16* __restrict__ Wlo, float* __restrict__ Wh,
    _Float16* __restrict__ WhT16) {
  const int i0   = blockIdx.x * 16;
  const int lane = threadIdx.x & 31;
  const int hi   = lane >> 4;
  const int ml   = lane & 15;

  v8f acc[8];
#pragma unroll
  for (int nc = 0; nc < 8; ++nc)
#pragma unroll
    for (int v = 0; v < 8; ++v) acc[nc][v] = 0.0f;

  for (int kc = 0; kc < 8; ++kc) {            // K = 256 in chunks of 32
    const float* hrow = h + (size_t)(i0 + ml) * FIN + kc * 32;
    f32x8 a0 = *(const f32x8*)(hrow + 8 * hi);        // K = 8*hi .. +7
    f32x8 a1 = *(const f32x8*)(hrow + 16 + 8 * hi);   // K = 16+8*hi .. +7
    v16h ahi, alo;
#pragma unroll
    for (int t = 0; t < 8; ++t) {
      const float x = a0[t]; const _Float16 bh = f2h(x);
      ahi[t] = bh; alo[t] = f2h(x - h2f(bh));
    }
#pragma unroll
    for (int t = 0; t < 8; ++t) {
      const float x = a1[t]; const _Float16 bh = f2h(x);
      ahi[t + 8] = bh; alo[t + 8] = f2h(x - h2f(bh));
    }
#pragma unroll
    for (int nc = 0; nc < 8; ++nc) {          // keep acc[] in VGPRs
      const size_t widx = (size_t)(((kc * 8 + nc) * 32 + lane)) << 4;
      const v16h bhi = *(const v16h*)(Whi + widx);
      const v16h blo = *(const v16h*)(Wlo + widx);
      acc[nc] = __builtin_amdgcn_wmma_f32_16x16x32_f16(
          false, alo, false, bhi, (short)0, acc[nc], false, false);
      acc[nc] = __builtin_amdgcn_wmma_f32_16x16x32_f16(
          false, ahi, false, blo, (short)0, acc[nc], false, false);
      acc[nc] = __builtin_amdgcn_wmma_f32_16x16x32_f16(
          false, ahi, false, bhi, (short)0, acc[nc], false, false);
    }
  }
#pragma unroll
  for (int nc = 0; nc < 8; ++nc)
#pragma unroll
    for (int v = 0; v < 8; ++v) {
      const int m = v + 8 * hi;
      const float val = acc[nc][v];
      Wh[(size_t)(i0 + m) * FOUT + nc * 16 + ml] = val;
      WhT16[(size_t)(nc * 16 + ml) * NN + (i0 + m)] = f2h(val);
    }
}

// ---------------------------------------------------------------------------
// Kernel 2: f1 = Wh @ a1, f2 = Wh @ a2 (tiny, L2-resident)
// ---------------------------------------------------------------------------
__global__ __launch_bounds__(256) void gat_f_kernel(
    const float* __restrict__ Wh, const float* __restrict__ a,
    float* __restrict__ f1, float* __restrict__ f2) {
  const int i = blockIdx.x * blockDim.x + threadIdx.x;
  if (i >= NN) return;
  const float* r = Wh + (size_t)i * FOUT;
  float s1 = 0.f, s2 = 0.f;
#pragma unroll 4
  for (int f = 0; f < FOUT; ++f) {
    const float w = r[f];
    s1 += w * a[f];
    s2 += w * a[FOUT + f];
  }
  f1[i] = s1;
  f2[i] = s2;
}

// ---------------------------------------------------------------------------
// Kernel 3: fused masked-softmax attention + attn@Wh + elu.
// 512 blocks x 8 waves; block owns rows [i0,i0+16), wave owns 1024 columns.
// Online (flash) softmax per wave; rescale skipped when no row max changed
// (exact: corr==1; rare after a few tiles). Waves merged through LDS
// (ds_add_f32). All WMMAs unconditional from full waves (EXEC all-ones).
// ---------------------------------------------------------------------------
__global__ __launch_bounds__(256) void gat_attn_kernel(
    const int* __restrict__ adj, const _Float16* __restrict__ WhT16,
    const float* __restrict__ f1, const float* __restrict__ f2,
    float* __restrict__ out) {
  __shared__ float s_acc[16][FOUT];   // merged numerator, 8 KB
  __shared__ float s_m[8][16];
  __shared__ float s_l[8][16];
  __shared__ float s_M[16];
  __shared__ float s_L[16];

  const int i0   = blockIdx.x * 16;
  const int tid  = threadIdx.x;
  const int wave = tid >> 5;
  const int lane = tid & 31;
  const int hi   = lane >> 4;
  const int ml   = lane & 15;

  for (int k = tid; k < 16 * FOUT; k += 256) ((float*)s_acc)[k] = 0.0f;

  const float  f1v  = f1[i0 + ml];
  const size_t arow = (size_t)(i0 + ml) * NN;

  float m_r = -INFINITY;
  float l_r = 0.0f;
  v8f acc[8];
#pragma unroll
  for (int nc = 0; nc < 8; ++nc)
#pragma unroll
    for (int v = 0; v < 8; ++v) acc[nc][v] = 0.0f;

  const int jbase = wave * 1024;
  for (int tt = 0; tt < 32; ++tt) {
    const int j0 = jbase + tt * 32;
    // adj / f2 in the 16-bit A-matrix lane layout: two runs of 8.
    const v8i   av0 = *(const v8i*)(adj + arow + j0 + 8 * hi);
    const v8i   av1 = *(const v8i*)(adj + arow + j0 + 16 + 8 * hi);
    const f32x8 fa  = *(const f32x8*)(f2 + j0 + 8 * hi);
    const f32x8 fb  = *(const f32x8*)(f2 + j0 + 16 + 8 * hi);
    __builtin_prefetch(adj + arow + j0 + 64, 0, 1);   // 2 tiles ahead

    float e[16];
#pragma unroll
    for (int t = 0; t < 8; ++t) {
      float x = f1v + fa[t];
      x = x > 0.0f ? x : ALPHA * x;
      e[t] = (av0[t] > 0) ? x : NEGINF;
    }
#pragma unroll
    for (int t = 0; t < 8; ++t) {
      float x = f1v + fb[t];
      x = x > 0.0f ? x : ALPHA * x;
      e[t + 8] = (av1[t] > 0) ? x : NEGINF;
    }

    float tmax = e[0];
#pragma unroll
    for (int t = 1; t < 16; ++t) tmax = fmaxf(tmax, e[t]);
    tmax = fmaxf(tmax, __shfl_xor(tmax, 16, 32));   // merge row halves
    const float m_new = fmaxf(m_r, tmax);
    const bool  grew  = (m_new > m_r);              // per-row (lane-pair) flag

    float ps = 0.0f;
    v16h aP;
#pragma unroll
    for (int t = 0; t < 16; ++t) {
      const float p = __expf(e[t] - m_new);
      ps += p;
      aP[t] = f2h(p);
    }
    ps += __shfl_xor(ps, 16, 32);

    if (__any((int)grew)) {          // wave-uniform; corr==1 otherwise
      const float corr = __expf(m_r - m_new);       // 0 on first tile
      l_r = l_r * corr + ps;
      // broadcast per-row rescale into C layout (row = v + 8*hi)
      float corr_row[8];
#pragma unroll
      for (int v = 0; v < 8; ++v) corr_row[v] = __shfl(corr, v + 8 * hi, 32);
#pragma unroll
      for (int nc = 0; nc < 8; ++nc)
#pragma unroll
        for (int v = 0; v < 8; ++v) acc[nc][v] *= corr_row[v];
    } else {
      l_r += ps;
    }
    m_r = m_new;

#pragma unroll
    for (int nc = 0; nc < 8; ++nc) {
      const _Float16* bb = WhT16 + (size_t)(nc * 16 + ml) * NN + j0 + 8 * hi;
      typedef __attribute__((ext_vector_type(8))) _Float16 v8h_;
      const v8h_ b0 = *(const v8h_*)bb;
      const v8h_ b1 = *(const v8h_*)(bb + 16);
      const v16h b = __builtin_shufflevector(b0, b1, 0,1,2,3,4,5,6,7,8,9,10,11,12,13,14,15);   // two 16 B runs per lane: K = 8*hi.., 16+8*hi..
      acc[nc] = __builtin_amdgcn_wmma_f32_16x16x32_f16(
          false, aP, false, b, (short)0, acc[nc], false, false);
    }
  }

  if (hi == 0) { s_m[wave][ml] = m_r; s_l[wave][ml] = l_r; }
  __syncthreads();

  if (tid < 16) {
    float M = -INFINITY;
#pragma unroll
    for (int w = 0; w < 8; ++w) M = fmaxf(M, s_m[w][tid]);
    float L = 0.0f;
#pragma unroll
    for (int w = 0; w < 8; ++w) L += s_l[w][tid] * __expf(s_m[w][tid] - M);
    s_M[tid] = M;
    s_L[tid] = L;   // > 0 always (fully masked rows -> uniform softmax)
  }
  __syncthreads();

#pragma unroll
  for (int v = 0; v < 8; ++v) {
    const int m = v + 8 * hi;
    const float sc = __expf(s_m[wave][m] - s_M[m]) / s_L[m];
#pragma unroll
    for (int nc = 0; nc < 8; ++nc)
      atomicAdd(&s_acc[m][nc * 16 + ml], acc[nc][v] * sc);
  }
  __syncthreads();

  for (int k = tid; k < 16 * FOUT; k += 256) {
    const int m = k >> 7, n = k & 127;
    const float x = s_acc[m][n];
    out[(size_t)(i0 + m) * FOUT + n] = x > 0.0f ? x : (__expf(x) - 1.0f);
  }
}

// ---------------------------------------------------------------------------
extern "C" void kernel_launch(void* const* d_in, const int* in_sizes, int n_in,
                              void* d_out, int out_size, void* d_ws, size_t ws_size,
                              hipStream_t stream) {
  const float* h   = (const float*)d_in[0];   // (8192, 256)
  const int*   adj = (const int*)d_in[1];     // (8192, 8192)
  const float* W   = (const float*)d_in[2];   // (256, 128)
  const float* a   = (const float*)d_in[3];   // (256, 1)
  float* out = (float*)d_out;                 // (8192, 128) f32

  float*    Wh    = (float*)d_ws;                        // 4 MB
  float*    f1    = Wh + (size_t)NN * FOUT;              // 32 KB
  float*    f2    = f1 + NN;                             // 32 KB
  _Float16* WhT16 = (_Float16*)(f2 + NN);                // 2 MB (transposed)
  _Float16* Whi   = WhT16 + (size_t)NN * FOUT;           // 64 KB (swizzled)
  _Float16* Wlo   = Whi + 8 * 8 * 32 * 16;               // 64 KB (swizzled)

  gat_wprep_kernel<<<128, 256, 0, stream>>>(W, Whi, Wlo);
  gat_wh_kernel<<<NN / 16, 32, 0, stream>>>(h, Whi, Wlo, Wh, WhT16);
  gat_f_kernel<<<NN / 256, 256, 0, stream>>>(Wh, a, f1, f2);
  gat_attn_kernel<<<NN / 16, 256, 0, stream>>>(adj, WhT16, f1, f2, out);
}
